// HunyuanImage3Model_86775519248874
// MI455X (gfx1250) — matrix operand order repaired and checked by reading, not compiled, not run on hardware
//
#include <hip/hip_runtime.h>
#include <hip/hip_bf16.h>

// ---------------- constants (match reference) ----------------
#define S_   1024
#define M_   1024
#define H_   16
#define D_   64
#define KV_  8
#define G_   2          // H/KV
#define FF_  3072
#define E_   64
#define TK_  8          // top-k
#define CAP_ 256        // 2*K*S*B/E
#define RMS_EPS_ 1e-6f
#define F32_EPS_ 1.1920929e-07f

typedef __attribute__((ext_vector_type(16))) __bf16 v16bf;
typedef __attribute__((ext_vector_type(8)))  float  v8f;

__device__ __forceinline__ unsigned short f2bf(float f) {
  unsigned u = __builtin_bit_cast(unsigned, f);
  unsigned r = u + 0x7FFFu + ((u >> 16) & 1u);   // round-to-nearest-even
  return (unsigned short)(r >> 16);
}
__device__ __forceinline__ uint2 pack4bf(float x, float y, float z, float w) {
  uint2 r;
  r.x = (unsigned)f2bf(x) | ((unsigned)f2bf(y) << 16);
  r.y = (unsigned)f2bf(z) | ((unsigned)f2bf(w) << 16);
  return r;
}

// =====================================================================
// Generic batched WMMA GEMM:  C[M,N] = A[M,K] * B   (f32 in, f32 out,
// bf16 convert while staging to LDS, v_wmma_f32_16x16x32_bf16 core).
// transB==0 : B row-major [K,N]   (B[k*ldb + n])
// transB==1 : B row-major [N,K]   (B[n*ldb + k], i.e. B^T)
// Block tile 128x256, 8 waves, wave tile 64x64 (4x4 16x16 frags).
// Double-buffered LDS, b128 global loads, b128 LDS fragment loads,
// global_prefetch for k+2 tiles. blockIdx.z = batch.
// =====================================================================
#define BM 128
#define BN 256
#define BK 32
#define LP 40   // LDS pitch in ushorts: 80B rows -> 16B-aligned, conflict-broken

__launch_bounds__(256)
__global__ void k_wmma_gemm(const float* __restrict__ A,
                            const float* __restrict__ B,
                            float* __restrict__ C,
                            int M, int N, int Kd,
                            int lda, int ldb, int ldc,
                            long long sA, long long sB, long long sC,
                            int transB)
{
  __shared__ __align__(16) unsigned short As[2][BM * LP];   // [m][k]
  __shared__ __align__(16) unsigned short Bs[2][BN * LP];   // [n][k]

  const int tid   = threadIdx.x;
  const int lane  = tid & 31;
  const int wave  = tid >> 5;       // 0..7
  const int waveM = wave >> 2;      // 0..1  -> 64 rows
  const int waveN = wave & 3;       // 0..3  -> 64 cols
  const int half  = lane >> 4;      // 0/1
  const int l16   = lane & 15;

  const int bm0 = blockIdx.y * BM;
  const int bn0 = blockIdx.x * BN;
  A += (long long)blockIdx.z * sA;
  B += (long long)blockIdx.z * sB;
  C += (long long)blockIdx.z * sC;

  v8f acc[4][4];
  for (int i = 0; i < 4; ++i)
    for (int j = 0; j < 4; ++j) acc[i][j] = v8f{};

  float4 a_reg[4];   // A staging: 4 rows x 4 floats
  float4 b_reg[8];   // B staging: 8 chunks x 4 floats

  // ---- guarded float4 load ----
  auto ld4 = [](const float* p, bool rowok, int c0, int lim) -> float4 {
    float4 v = make_float4(0.f, 0.f, 0.f, 0.f);
    if (rowok) {
      if (c0 + 3 < lim) v = *(const float4*)p;
      else {
        if (c0     < lim) v.x = p[0];
        if (c0 + 1 < lim) v.y = p[1];
        if (c0 + 2 < lim) v.z = p[2];
        if (c0 + 3 < lim) v.w = p[3];
      }
    }
    return v;
  };

  auto g_load = [&](int k0) {
    // A tile 128x32
    for (int it = 0; it < 4; ++it) {
      int r = (tid >> 3) + 32 * it, cgrp = (tid & 7) * 4;
      int gm = bm0 + r, gk = k0 + cgrp;
      a_reg[it] = ld4(A + (long long)gm * lda + gk, gm < M, gk, Kd);
    }
    if (transB == 0) {     // B tile 32(k) x 256(n)
      for (int it = 0; it < 8; ++it) {
        int r = (tid >> 6) + 4 * it, cgrp = (tid & 63) * 4;
        int gk = k0 + r, gn = bn0 + cgrp;
        b_reg[it] = ld4(B + (long long)gk * ldb + gn, gk < Kd, gn, N);
      }
    } else {               // B tile 256(n) x 32(k)
      for (int it = 0; it < 8; ++it) {
        int r = (tid >> 3) + 32 * it, cgrp = (tid & 7) * 4;
        int gn = bn0 + r, gk = k0 + cgrp;
        b_reg[it] = ld4(B + (long long)gn * ldb + gk, gn < N, gk, Kd);
      }
    }
  };

  auto s_store = [&](int buf) {
    for (int it = 0; it < 4; ++it) {
      int r = (tid >> 3) + 32 * it, cgrp = (tid & 7) * 4;
      *(uint2*)&As[buf][r * LP + cgrp] =
          pack4bf(a_reg[it].x, a_reg[it].y, a_reg[it].z, a_reg[it].w);
    }
    if (transB == 0) {     // scatter n-major into Bs[n][k]
      for (int it = 0; it < 8; ++it) {
        int r = (tid >> 6) + 4 * it, cgrp = (tid & 63) * 4;
        Bs[buf][(cgrp + 0) * LP + r] = f2bf(b_reg[it].x);
        Bs[buf][(cgrp + 1) * LP + r] = f2bf(b_reg[it].y);
        Bs[buf][(cgrp + 2) * LP + r] = f2bf(b_reg[it].z);
        Bs[buf][(cgrp + 3) * LP + r] = f2bf(b_reg[it].w);
      }
    } else {
      for (int it = 0; it < 8; ++it) {
        int r = (tid >> 3) + 32 * it, cgrp = (tid & 7) * 4;
        *(uint2*)&Bs[buf][r * LP + cgrp] =
            pack4bf(b_reg[it].x, b_reg[it].y, b_reg[it].z, b_reg[it].w);
      }
    }
  };

  // prologue
  g_load(0);
  s_store(0);
  int cur = 0;
  __syncthreads();

  for (int k0 = 0; k0 < Kd; k0 += BK) {
    int nk = k0 + BK;
    if (nk < Kd) {
      g_load(nk);
      // prefetch k+2 tile toward L2 (global_prefetch_b8)
      if (nk + BK < Kd) {
        __builtin_prefetch(A + (long long)(bm0 + (tid >> 3)) * lda + (nk + BK) + (tid & 7) * 4, 0, 0);
        if (transB == 0)
          __builtin_prefetch(B + (long long)(nk + BK + (tid >> 6)) * ldb + bn0 + (tid & 63) * 4, 0, 0);
        else
          __builtin_prefetch(B + (long long)(bn0 + (tid >> 3)) * ldb + (nk + BK) + (tid & 7) * 4, 0, 0);
      }
    }

    // ---- fragments from LDS (ds_load_b128) ----
    union Frag { v16bf v; uint4 q[2]; };
    Frag afr[4], bfr[4];
    for (int fi = 0; fi < 4; ++fi) {
      int m = waveM * 64 + fi * 16 + l16;
      const unsigned short* p = &As[cur][m * LP];
      afr[fi].q[0] = *(const uint4*)(p + half * 8);        // K = half*8 + 0..7
      afr[fi].q[1] = *(const uint4*)(p + 16 + half * 8);   // K = 16 + half*8 + 0..7
    }
    for (int fj = 0; fj < 4; ++fj) {
      int n = waveN * 64 + fj * 16 + l16;
      const unsigned short* p = &Bs[cur][n * LP + half * 8];
      bfr[fj].q[0] = *(const uint4*)(p);
      bfr[fj].q[1] = *(const uint4*)(p + 16);
    }
    for (int fi = 0; fi < 4; ++fi)
      for (int fj = 0; fj < 4; ++fj)
        acc[fi][fj] = __builtin_amdgcn_wmma_f32_16x16x32_bf16(
            false, afr[fi].v, false, bfr[fj].v, (short)0, acc[fi][fj], false, false);

    if (nk < Kd) s_store(cur ^ 1);
    __syncthreads();
    cur ^= 1;
  }

  // ---- epilogue: C/D layout VGPR r -> M = base + half*8 + r ----
  for (int fi = 0; fi < 4; ++fi)
    for (int fj = 0; fj < 4; ++fj) {
      int n = bn0 + waveN * 64 + fj * 16 + l16;
      if (n >= N) continue;
      int mbase = bm0 + waveM * 64 + fi * 16 + half * 8;
      for (int r = 0; r < 8; ++r) {
        int m = mbase + r;
        if (m < M) C[(long long)m * ldc + n] = acc[fi][fj][r];
      }
    }
}

// =====================================================================
// RMSNorm: y = w * x * rsqrt(mean(x^2)+eps)   one block per row
// =====================================================================
__global__ void k_rmsnorm(const float* __restrict__ x, const float* __restrict__ w,
                          float* __restrict__ y, int cols)
{
  int row = blockIdx.x, tid = threadIdx.x;
  const float* xr = x + (long long)row * cols;
  float s = 0.f;
  for (int c = tid; c < cols; c += 256) { float v = xr[c]; s += v * v; }
  __shared__ float red[256];
  red[tid] = s; __syncthreads();
  for (int off = 128; off > 0; off >>= 1) { if (tid < off) red[tid] += red[tid + off]; __syncthreads(); }
  float inv = rsqrtf(red[0] / (float)cols + RMS_EPS_);
  for (int c = tid; c < cols; c += 256) y[(long long)row * cols + c] = w[c] * xr[c] * inv;
}

// =====================================================================
// QKV split + RoPE + QK-RMSNorm -> Q/K/V as [H][S][D] (K,V repeated G)
// grid (S, KV), block D=64
// =====================================================================
__global__ void k_qkv_rope(const float* __restrict__ qkv,
                           const float* __restrict__ cosb, const float* __restrict__ sinb,
                           const float* __restrict__ qn, const float* __restrict__ kn,
                           float* __restrict__ Q, float* __restrict__ Kr, float* __restrict__ Vr)
{
  int s = blockIdx.x, kv = blockIdx.y, d = threadIdx.x;
  const float* base = qkv + (long long)s * ((H_ + 2 * KV_) * D_) + kv * (G_ + 2) * D_;
  float c  = cosb[s * D_ + d];
  float sn = sinb[s * D_ + d];

  __shared__ float sh[4][D_];
  for (int slot = 0; slot < 4; ++slot) sh[slot][d] = base[slot * D_ + d];
  __syncthreads();

  float out[4];
  for (int slot = 0; slot < 3; ++slot) {   // q0, q1, k get RoPE
    float xv  = sh[slot][d];
    float rot = (d < D_ / 2) ? -sh[slot][d + D_ / 2] : sh[slot][d - D_ / 2];
    out[slot] = xv * c + rot * sn;
  }
  out[3] = sh[3][d];

  __shared__ float red[D_];
  for (int slot = 0; slot < 3; ++slot) {
    red[d] = out[slot] * out[slot]; __syncthreads();
    for (int off = 32; off > 0; off >>= 1) { if (d < off) red[d] += red[d + off]; __syncthreads(); }
    float inv = rsqrtf(red[0] / (float)D_ + RMS_EPS_);
    float val = out[slot] * inv * ((slot < 2) ? qn[d] : kn[d]);
    if (slot < 2) {
      int h = kv * G_ + slot;
      Q[((long long)h * S_ + s) * D_ + d] = val;
    } else {
      for (int g = 0; g < G_; ++g)
        Kr[((long long)(kv * G_ + g) * S_ + s) * D_ + d] = val;
    }
    __syncthreads();
  }
  for (int g = 0; g < G_; ++g)
    Vr[((long long)(kv * G_ + g) * S_ + s) * D_ + d] = out[3];
}

// =====================================================================
// Causal softmax in place over scores [H][S][S], scale 1/sqrt(D)
// grid (S, H), block 256
// =====================================================================
__global__ void k_softmax_causal(float* __restrict__ sc)
{
  int q = blockIdx.x, h = blockIdx.y, tid = threadIdx.x;
  float* row = sc + ((long long)h * S_ + q) * (long long)S_;
  const float scale = 0.125f;   // 1/sqrt(64)
  __shared__ float red[256];

  float mx = -1e30f;
  for (int j = tid; j < S_; j += 256) if (j <= q) mx = fmaxf(mx, row[j] * scale);
  red[tid] = mx; __syncthreads();
  for (int off = 128; off > 0; off >>= 1) { if (tid < off) red[tid] = fmaxf(red[tid], red[tid + off]); __syncthreads(); }
  float m = red[0]; __syncthreads();

  float sum = 0.f;
  for (int j = tid; j < S_; j += 256) {
    if (j <= q) { float e = __expf(row[j] * scale - m); row[j] = e; sum += e; }
    else row[j] = 0.f;
  }
  red[tid] = sum; __syncthreads();
  for (int off = 128; off > 0; off >>= 1) { if (tid < off) red[tid] += red[tid + off]; __syncthreads(); }
  float inv = 1.f / red[0];
  for (int j = tid; j <= q; j += 256) row[j] *= inv;
}

// =====================================================================
// SwiGLU: y[r,f] = gu[r,f] * silu(gu[r,FF+f]) ; gu row stride 2*ff
// =====================================================================
__global__ void k_silu_mul(const float* __restrict__ gu, float* __restrict__ y,
                           long long total, int ff)
{
  long long i = (long long)blockIdx.x * blockDim.x + threadIdx.x;
  long long stride = (long long)gridDim.x * blockDim.x;
  for (; i < total; i += stride) {
    long long r = i / ff; int f = (int)(i % ff);
    float x1 = gu[r * (2LL * ff) + f];
    float x2 = gu[r * (2LL * ff) + ff + f];
    y[i] = x1 * (x2 / (1.f + __expf(-x2)));
  }
}

// out = a + b (elementwise)
__global__ void k_add2(const float* __restrict__ a, const float* __restrict__ b,
                       float* __restrict__ o, long long n)
{
  long long i = (long long)blockIdx.x * blockDim.x + threadIdx.x;
  long long stride = (long long)gridDim.x * blockDim.x;
  for (; i < n; i += stride) o[i] = a[i] + b[i];
}

// =====================================================================
// Router: softmax over E, top-8, normalized combine weights.
// One block (64 threads) per token; shared-memory softmax, serial top-k.
// =====================================================================
__global__ void k_gate_topk(const float* __restrict__ logits,
                            int* __restrict__ topi, float* __restrict__ topw)
{
  int s = blockIdx.x, t = threadIdx.x;    // t in [0,64)
  __shared__ float g[E_];
  __shared__ float c[E_];
  __shared__ float red[E_];

  float v = logits[(long long)s * E_ + t];
  red[t] = v; __syncthreads();
  for (int off = 32; off > 0; off >>= 1) { if (t < off) red[t] = fmaxf(red[t], red[t + off]); __syncthreads(); }
  float mx = red[0]; __syncthreads();
  float e = __expf(v - mx);
  red[t] = e; __syncthreads();
  for (int off = 32; off > 0; off >>= 1) { if (t < off) red[t] += red[t + off]; __syncthreads(); }
  float sum = red[0];
  g[t] = e / sum;
  c[t] = g[t];
  __syncthreads();

  if (t == 0) {
    int idxs[TK_]; float sv = 0.f;
    for (int k = 0; k < TK_; ++k) {
      int best = 0; float bv = -1.f;
      for (int i = 0; i < E_; ++i) if (c[i] > bv) { bv = c[i]; best = i; }
      idxs[k] = best; sv += g[best]; c[best] = -2.f;
    }
    if (sv < F32_EPS_) sv = F32_EPS_;
    for (int k = 0; k < TK_; ++k) {
      topi[s * TK_ + k] = idxs[k];
      topw[s * TK_ + k] = g[idxs[k]] / sv;
    }
  }
}

// =====================================================================
// Deterministic capacity assignment (matches reference cumsum order:
// k-major over tokens). 1 block, E threads — each expert scans serially.
// =====================================================================
__global__ void k_assign(const int* __restrict__ topi, const float* __restrict__ topw,
                         int* __restrict__ slot_token,
                         int* __restrict__ tok_e, int* __restrict__ tok_c,
                         float* __restrict__ tok_w)
{
  int e = threadIdx.x;
  if (e >= E_) return;
  for (int c = 0; c < CAP_; ++c) slot_token[e * CAP_ + c] = -1;
  int cnt = 0;
  for (int k = 0; k < TK_; ++k)
    for (int s = 0; s < S_; ++s) {
      int idx = s * TK_ + k;
      if (topi[idx] == e) {
        if (cnt < CAP_) {
          slot_token[e * CAP_ + cnt] = s;
          tok_e[idx] = e; tok_c[idx] = cnt; tok_w[idx] = topw[idx];
        } else {
          tok_e[idx] = e; tok_c[idx] = -1; tok_w[idx] = 0.f;
        }
        ++cnt;
      }
    }
}

// gather dispatched rows: X_e[e*CAP+c, :] = x[token] or 0
__global__ void k_build_dispatch(const int* __restrict__ slot_token,
                                 const float* __restrict__ x, float* __restrict__ Xe)
{
  int row = blockIdx.x;                 // 0..E*CAP-1
  int t = slot_token[row];
  float* dst = Xe + (long long)row * M_;
  if (t >= 0) {
    const float* src = x + (long long)t * M_;
    for (int m = threadIdx.x; m < M_; m += 256) dst[m] = src[m];
  } else {
    for (int m = threadIdx.x; m < M_; m += 256) dst[m] = 0.f;
  }
}

// out[s] = h[s] + shared[s] + sum_k w_k * eo[e_k, c_k]
__global__ void k_moe_combine(const float* __restrict__ h, const float* __restrict__ sh,
                              const float* __restrict__ eo,
                              const int* __restrict__ tok_e, const int* __restrict__ tok_c,
                              const float* __restrict__ tok_w, float* __restrict__ out)
{
  int s = blockIdx.x;
  int e[TK_], c[TK_]; float w[TK_];
  for (int k = 0; k < TK_; ++k) {
    e[k] = tok_e[s * TK_ + k]; c[k] = tok_c[s * TK_ + k]; w[k] = tok_w[s * TK_ + k];
  }
  for (int m = threadIdx.x; m < M_; m += 256) {
    float acc = h[(long long)s * M_ + m] + sh[(long long)s * M_ + m];
    for (int k = 0; k < TK_; ++k)
      if (c[k] >= 0)
        acc += w[k] * eo[((long long)e[k] * CAP_ + c[k]) * M_ + m];
    out[(long long)s * M_ + m] = acc;
  }
}

// =====================================================================
// host-side launcher
// =====================================================================
extern "C" void kernel_launch(void* const* d_in, const int* in_sizes, int n_in,
                              void* d_out, int out_size, void* d_ws, size_t ws_size,
                              hipStream_t stream)
{
  const float* hidden    = (const float*)d_in[0];
  const float* cosb      = (const float*)d_in[1];
  const float* sinb      = (const float*)d_in[2];
  const float* ln1_w     = (const float*)d_in[3];
  const float* ln2_w     = (const float*)d_in[4];
  const float* qkv_w     = (const float*)d_in[5];
  const float* o_w       = (const float*)d_in[6];
  const float* qn_w      = (const float*)d_in[7];
  const float* kn_w      = (const float*)d_in[8];
  const float* sh_gu_w   = (const float*)d_in[9];
  const float* sh_dn_w   = (const float*)d_in[10];
  const float* gate_w    = (const float*)d_in[11];
  const float* ex_gu_w   = (const float*)d_in[12];
  const float* ex_dn_w   = (const float*)d_in[13];
  float* out = (float*)d_out;

  // bump allocator over d_ws
  char* wp = (char*)d_ws;
  auto alloc = [&](size_t bytes) -> void* {
    void* p = (void*)wp;
    wp += (bytes + 255) & ~(size_t)255;
    return p;
  };
  float* hn       = (float*)alloc((size_t)S_ * M_ * 4);
  float* qkv      = (float*)alloc((size_t)S_ * (H_ + 2 * KV_) * D_ * 4);
  float* Q        = (float*)alloc((size_t)H_ * S_ * D_ * 4);
  float* Kr       = (float*)alloc((size_t)H_ * S_ * D_ * 4);
  float* Vr       = (float*)alloc((size_t)H_ * S_ * D_ * 4);
  float* scores   = (float*)alloc((size_t)H_ * S_ * S_ * 4);
  float* ao       = (float*)alloc((size_t)S_ * H_ * D_ * 4);
  float* attn_out = (float*)alloc((size_t)S_ * M_ * 4);
  float* hbuf     = (float*)alloc((size_t)S_ * M_ * 4);
  float* h2       = (float*)alloc((size_t)S_ * M_ * 4);
  float* gu_sh    = (float*)alloc((size_t)S_ * 2 * FF_ * 4);
  float* y_sh     = (float*)alloc((size_t)S_ * FF_ * 4);
  float* shared_o = (float*)alloc((size_t)S_ * M_ * 4);
  float* logits   = (float*)alloc((size_t)S_ * E_ * 4);
  int*   topi     = (int*)  alloc((size_t)S_ * TK_ * 4);
  float* topw     = (float*)alloc((size_t)S_ * TK_ * 4);
  int*   slot_tok = (int*)  alloc((size_t)E_ * CAP_ * 4);
  int*   tok_e    = (int*)  alloc((size_t)S_ * TK_ * 4);
  int*   tok_c    = (int*)  alloc((size_t)S_ * TK_ * 4);
  float* tok_w    = (float*)alloc((size_t)S_ * TK_ * 4);
  float* Xe       = (float*)alloc((size_t)E_ * CAP_ * M_ * 4);
  float* gu_e     = (float*)alloc((size_t)E_ * CAP_ * 2 * FF_ * 4);
  float* y_e      = (float*)alloc((size_t)E_ * CAP_ * FF_ * 4);
  float* eo       = (float*)alloc((size_t)E_ * CAP_ * M_ * 4);
  (void)ws_size; (void)n_in; (void)in_sizes; (void)out_size;

  const int QKVN = (H_ + 2 * KV_) * D_;   // 2048
  auto gx = [](int n) { return (n + BN - 1) / BN; };
  auto gy = [](int m) { return (m + BM - 1) / BM; };

  // 1) ln1
  k_rmsnorm<<<S_, 256, 0, stream>>>(hidden, ln1_w, hn, M_);

  // 2) qkv = hn @ qkv_w   [1024 x 2048 x 1024]
  k_wmma_gemm<<<dim3(gx(QKVN), gy(S_), 1), 256, 0, stream>>>(
      hn, qkv_w, qkv, S_, QKVN, M_, M_, QKVN, QKVN, 0, 0, 0, 0);

  // 3) RoPE + qk-norm + GQA repeat
  k_qkv_rope<<<dim3(S_, KV_), D_, 0, stream>>>(qkv, cosb, sinb, qn_w, kn_w, Q, Kr, Vr);

  // 4) scores[h] = Q[h] @ K[h]^T   batch H, [S x S x D]
  k_wmma_gemm<<<dim3(gx(S_), gy(S_), H_), 256, 0, stream>>>(
      Q, Kr, scores, S_, S_, D_, D_, D_, S_,
      (long long)S_ * D_, (long long)S_ * D_, (long long)S_ * S_, 1);

  // 5) causal softmax
  k_softmax_causal<<<dim3(S_, H_), 256, 0, stream>>>(scores);

  // 6) ao[h] = P[h] @ V[h]  -> written interleaved as [S, H*D]
  k_wmma_gemm<<<dim3(gx(D_), gy(S_), H_), 256, 0, stream>>>(
      scores, Vr, ao, S_, D_, S_, S_, D_, H_ * D_,
      (long long)S_ * S_, (long long)S_ * D_, (long long)D_, 0);

  // 7) attn_out = ao @ o_w
  k_wmma_gemm<<<dim3(gx(M_), gy(S_), 1), 256, 0, stream>>>(
      ao, o_w, attn_out, S_, M_, H_ * D_, H_ * D_, M_, M_, 0, 0, 0, 0);

  // 8) h = hidden + attn_out
  k_add2<<<1024, 256, 0, stream>>>(hidden, attn_out, hbuf, (long long)S_ * M_);

  // 9) ln2
  k_rmsnorm<<<S_, 256, 0, stream>>>(hbuf, ln2_w, h2, M_);

  // 10) shared MLP gate/up
  k_wmma_gemm<<<dim3(gx(2 * FF_), gy(S_), 1), 256, 0, stream>>>(
      h2, sh_gu_w, gu_sh, S_, 2 * FF_, M_, M_, 2 * FF_, 2 * FF_, 0, 0, 0, 0);
  // 11) swiglu
  k_silu_mul<<<2048, 256, 0, stream>>>(gu_sh, y_sh, (long long)S_ * FF_, FF_);
  // 12) shared down
  k_wmma_gemm<<<dim3(gx(M_), gy(S_), 1), 256, 0, stream>>>(
      y_sh, sh_dn_w, shared_o, S_, M_, FF_, FF_, M_, M_, 0, 0, 0, 0);

  // 13) router logits = h2 @ gate_w   [1024 x 64 x 1024]
  k_wmma_gemm<<<dim3(gx(E_), gy(S_), 1), 256, 0, stream>>>(
      h2, gate_w, logits, S_, E_, M_, M_, E_, E_, 0, 0, 0, 0);

  // 14) softmax + top-8 per token (one 64-thread block per token)
  k_gate_topk<<<S_, E_, 0, stream>>>(logits, topi, topw);
  // 15) capacity assignment (deterministic, matches reference order)
  k_assign<<<1, E_, 0, stream>>>(topi, topw, slot_tok, tok_e, tok_c, tok_w);
  // 16) gather dispatched tokens
  k_build_dispatch<<<E_ * CAP_, 256, 0, stream>>>(slot_tok, h2, Xe);

  // 17) expert gate/up : batch E, [CAP x 2FF x M]
  k_wmma_gemm<<<dim3(gx(2 * FF_), gy(CAP_), E_), 256, 0, stream>>>(
      Xe, ex_gu_w, gu_e, CAP_, 2 * FF_, M_, M_, 2 * FF_, 2 * FF_,
      (long long)CAP_ * M_, (long long)M_ * 2 * FF_, (long long)CAP_ * 2 * FF_, 0);
  // 18) expert swiglu
  k_silu_mul<<<4096, 256, 0, stream>>>(gu_e, y_e, (long long)E_ * CAP_ * FF_, FF_);
  // 19) expert down : batch E, [CAP x M x FF]
  k_wmma_gemm<<<dim3(gx(M_), gy(CAP_), E_), 256, 0, stream>>>(
      y_e, ex_dn_w, eo, CAP_, M_, FF_, FF_, M_, M_,
      (long long)CAP_ * FF_, (long long)FF_ * M_, (long long)CAP_ * M_, 0);

  // 20) combine: out = h + shared + sum_k w*eo
  k_moe_combine<<<S_, 256, 0, stream>>>(hbuf, shared_o, eo, tok_e, tok_c, tok_w, out);
}
